// NerfModel_80968723464896
// MI455X (gfx1250) — matrix operand order repaired and checked by reading, not compiled, not run on hardware
//
#include <hip/hip_runtime.h>
#include <math.h>

// ---------------------------------------------------------------------------
// NeRF forward for MI455X (gfx1250).  Core compute = f16 WMMA GEMMs
// (v_wmma_f32_16x16x32_f16), f32 accumulate, bias+ReLU fused.
// GEMM is register-blocked 2x4 (32 rows x 64 cols per wave, 8 accumulators)
// so each K-step issues 8 independent WMMAs per 6 b128-load pairs.
// ---------------------------------------------------------------------------

typedef __attribute__((ext_vector_type(8)))  _Float16 v8h;
typedef __attribute__((ext_vector_type(16))) _Float16 v16h;
typedef __attribute__((ext_vector_type(8)))  float    v8f;

#define NUM_COARSE 64
#define NUM_FINE   128
#define NTOT       192           // 64 + 128
#define POINT_ENC  63            // padded to 64
#define VIEW_ENC   27            // padded to 32

// ---------------------------------------------------------------------------
// Weight prep: f32 [K x OUT] row-major  ->  f16 [OUT x Kpad], zero padded K.
// ---------------------------------------------------------------------------
__global__ void wt_prep_kernel(const float* __restrict__ W, _Float16* __restrict__ Wt,
                               int K, int OUT, int Kpad) {
  long idx = (long)blockIdx.x * blockDim.x + threadIdx.x;
  long total = (long)OUT * Kpad;
  if (idx >= total) return;
  int o = (int)(idx / Kpad);
  int k = (int)(idx % Kpad);
  float v = (k < K) ? W[(long)k * OUT + o] : 0.0f;
  Wt[idx] = (_Float16)v;
}

// ---------------------------------------------------------------------------
// WMMA GEMM: Y[N,OUT] = act(X[N,K] @ Wt^T + bias).
//   X  : f16, row stride ldx, K multiple of 32
//   Wt : f16, [OUT x ldw] (transposed weights), ldw = Kpad
//   Y  : f16, row stride ldy
// One wave computes a 32-row x 64-col block: 2 row tiles x 4 col tiles,
// 8 f32 accumulators.  Requires N % 32 == 0 (true for all call sites).
// Fragment layouts follow CDNA5 ISA 7.12.2 (16-bit A / B / 32-bit C-D).
// ---------------------------------------------------------------------------
__device__ __forceinline__ v16h ld_afrag(const _Float16* p) {
  // p already includes row base + half*8; covers K {0..7, 16..23} relative
  v8h lo = *(const v8h*)(p);
  v8h hi = *(const v8h*)(p + 16);
  return __builtin_shufflevector(lo, hi, 0,1,2,3,4,5,6,7,8,9,10,11,12,13,14,15);
}
__device__ __forceinline__ v16h ld_bfrag(const _Float16* p) {
  // p already includes col base + half*8; covers K {0..7, 16..23} relative
  v8h lo = *(const v8h*)(p);
  v8h hi = *(const v8h*)(p + 16);
  return __builtin_shufflevector(lo, hi, 0,1,2,3,4,5,6,7,8,9,10,11,12,13,14,15);
}
#define WMMA_F16(a, b, c) \
  __builtin_amdgcn_wmma_f32_16x16x32_f16(false, (a), false, (b), (short)0, (c), false, false)

__global__ void gemm_wmma_kernel(const _Float16* __restrict__ X, int ldx,
                                 const _Float16* __restrict__ Wt, int ldw,
                                 const float* __restrict__ bias,
                                 _Float16* __restrict__ Y, int ldy,
                                 long N, int K, int OUT, int relu) {
  int wave = threadIdx.x >> 5;
  int lane = threadIdx.x & 31;
  long rowBlk = (long)blockIdx.x * (blockDim.x >> 5) + wave;   // 32 rows per wave
  long row0 = rowBlk * 32;
  if (row0 >= N) return;                       // uniform per wave -> EXEC stays all-1s

  int half = lane >> 4;                        // 0: lanes 0-15, 1: lanes 16-31
  int l15  = lane & 15;

  const _Float16* Arow0 = X + (row0 + l15) * (long)ldx + half * 8;
  const _Float16* Arow1 = Arow0 + 16L * ldx;
  int  bkb = half * 8;
  long cs  = 16L * ldw;                        // column-tile stride in Wt

  int nGrp = OUT >> 6;                         // groups of 4 col tiles
  for (int g = 0; g < nGrp; ++g) {
    const _Float16* Bp = Wt + ((long)g * 64 + l15) * (long)ldw + bkb;
    v8f a00 = {}, a01 = {}, a02 = {}, a03 = {};
    v8f a10 = {}, a11 = {}, a12 = {}, a13 = {};
    for (int k0 = 0; k0 < K; k0 += 32) {
      v16h fa0 = ld_afrag(Arow0 + k0);
      v16h fa1 = ld_afrag(Arow1 + k0);
      v16h fb0 = ld_bfrag(Bp + k0);
      v16h fb1 = ld_bfrag(Bp + cs + k0);
      v16h fb2 = ld_bfrag(Bp + 2 * cs + k0);
      v16h fb3 = ld_bfrag(Bp + 3 * cs + k0);
      a00 = WMMA_F16(fa0, fb0, a00);
      a01 = WMMA_F16(fa0, fb1, a01);
      a02 = WMMA_F16(fa0, fb2, a02);
      a03 = WMMA_F16(fa0, fb3, a03);
      a10 = WMMA_F16(fa1, fb0, a10);
      a11 = WMMA_F16(fa1, fb1, a11);
      a12 = WMMA_F16(fa1, fb2, a12);
      a13 = WMMA_F16(fa1, fb3, a13);
    }
    // C/D layout: VGPR r -> (M = r + half*8, N = l15)
    auto store_tile = [&](const v8f& acc, long r0, int colbase) {
      int col = colbase + l15;
      float bv = bias[col];
      #pragma unroll
      for (int r = 0; r < 8; ++r) {
        float y = acc[r] + bv;
        if (relu) y = y > 0.0f ? y : 0.0f;
        Y[(r0 + r + half * 8) * (long)ldy + col] = (_Float16)y;
      }
    };
    int cb = g * 64;
    store_tile(a00, row0,      cb);
    store_tile(a01, row0,      cb + 16);
    store_tile(a02, row0,      cb + 32);
    store_tile(a03, row0,      cb + 48);
    store_tile(a10, row0 + 16, cb);
    store_tile(a11, row0 + 16, cb + 16);
    store_tile(a12, row0 + 16, cb + 32);
    store_tile(a13, row0 + 16, cb + 48);
  }
}

// ---------------------------------------------------------------------------
// Positional encoding (min_deg=0, max_deg=10) -> 63 features (+1 zero pad).
// enc = [x, sin(2^d * x) d-major, cos(2^d * x) d-major]
// ---------------------------------------------------------------------------
__device__ inline void posenc10(float x, float y, float z, _Float16* out) {
  out[0] = (_Float16)x; out[1] = (_Float16)y; out[2] = (_Float16)z;
  float v[3] = {x, y, z};
  float s = 1.0f;
  #pragma unroll
  for (int d = 0; d < 10; ++d) {
    #pragma unroll
    for (int c = 0; c < 3; ++c) {
      float a = v[c] * s;
      out[3  + d * 3 + c] = (_Float16)__sinf(a);
      out[33 + d * 3 + c] = (_Float16)__cosf(a);
    }
    s *= 2.0f;
  }
  out[63] = (_Float16)0.0f;
}

// Coarse: deterministic stratified z, sample points, posenc.
__global__ void gen_coarse_kernel(const float* __restrict__ origins,
                                  const float* __restrict__ dirs,
                                  _Float16* __restrict__ enc, float* __restrict__ zc,
                                  int B) {
  long idx = (long)blockIdx.x * blockDim.x + threadIdx.x;
  if (idx >= (long)B * NUM_COARSE) return;
  int ray = (int)(idx >> 6);
  int i   = (int)(idx & 63);
  float t  = (float)i / (float)(NUM_COARSE - 1);
  float zv = 2.0f * (1.0f - t) + 6.0f * t;
  zc[idx] = zv;
  float px = origins[ray*3+0] + zv * dirs[ray*3+0];
  float py = origins[ray*3+1] + zv * dirs[ray*3+1];
  float pz = origins[ray*3+2] + zv * dirs[ray*3+2];
  posenc10(px, py, pz, enc + idx * 64);
}

// Fine: posenc of merged z samples.
__global__ void gen_fine_kernel(const float* __restrict__ origins,
                                const float* __restrict__ dirs,
                                const float* __restrict__ zf,
                                _Float16* __restrict__ enc, int B) {
  long idx = (long)blockIdx.x * blockDim.x + threadIdx.x;
  if (idx >= (long)B * NTOT) return;
  int ray = (int)(idx / NTOT);
  float zv = zf[idx];
  float px = origins[ray*3+0] + zv * dirs[ray*3+0];
  float py = origins[ray*3+1] + zv * dirs[ray*3+1];
  float pz = origins[ray*3+2] + zv * dirs[ray*3+2];
  posenc10(px, py, pz, enc + idx * 64);
}

// View-dir posenc (deg_view=4) -> 27 features, padded to 32 (f32).
__global__ void gen_vdenc_kernel(const float* __restrict__ vd,
                                 float* __restrict__ out, int B) {
  int ray = blockIdx.x * blockDim.x + threadIdx.x;
  if (ray >= B) return;
  float v[3] = {vd[ray*3+0], vd[ray*3+1], vd[ray*3+2]};
  float* o = out + ray * 32;
  o[0] = v[0]; o[1] = v[1]; o[2] = v[2];
  float s = 1.0f;
  for (int d = 0; d < 4; ++d) {
    for (int c = 0; c < 3; ++c) {
      float a = v[c] * s;
      o[3  + d * 3 + c] = __sinf(a);
      o[15 + d * 3 + c] = __cosf(a);
    }
    s *= 2.0f;
  }
  for (int j = 27; j < 32; ++j) o[j] = 0.0f;
}

// Fill skip-concat columns 256..319 of the layer-5 input (cols 256..318 = enc, 319 = 0).
__global__ void skip_fill_kernel(_Float16* __restrict__ buf,
                                 const _Float16* __restrict__ enc, long N) {
  long idx = (long)blockIdx.x * blockDim.x + threadIdx.x;
  if (idx >= N * 64) return;
  long n = idx >> 6;
  int  j = (int)(idx & 63);
  buf[n * 320 + 256 + j] = (j < 63) ? enc[n * 64 + j] : (_Float16)0.0f;
}

// Fill cond-concat columns 256..287 (256..282 = vdirs_enc for this ray, rest 0).
__global__ void cond_fill_kernel(_Float16* __restrict__ buf,
                                 const float* __restrict__ vdenc, long N, int S) {
  long idx = (long)blockIdx.x * blockDim.x + threadIdx.x;
  if (idx >= N * 32) return;
  long n = idx >> 5;
  int  j = (int)(idx & 31);
  int ray = (int)(n / S);
  float v = (j < 27) ? vdenc[ray * 32 + j] : 0.0f;
  buf[n * 320 + 256 + j] = (_Float16)v;
}

// sigma head: raw_sigma[n] = x[n,:] . w_sigma + b_sigma   (wave per row)
__global__ void sigma_dot_kernel(const _Float16* __restrict__ X, int ldx,
                                 const float* __restrict__ w, const float* __restrict__ b,
                                 float* __restrict__ out, long N) {
  int wave = threadIdx.x >> 5, lane = threadIdx.x & 31;
  long row = (long)blockIdx.x * 8 + wave;
  if (row >= N) return;
  const _Float16* xr = X + row * (long)ldx;
  float s = 0.0f;
  for (int k = lane; k < 256; k += 32) s += (float)xr[k] * w[k];
  for (int off = 16; off; off >>= 1) s += __shfl_down(s, off, 32);
  if (lane == 0) out[row] = s + b[0];
}

// rgb head: sigmoid(h . w_rgb + b_rgb)   (wave per row, 3 dot products of 128)
__global__ void rgb_dot_kernel(const _Float16* __restrict__ X, int ldx,
                               const float* __restrict__ w, const float* __restrict__ b,
                               float* __restrict__ out, long N) {
  int wave = threadIdx.x >> 5, lane = threadIdx.x & 31;
  long row = (long)blockIdx.x * 8 + wave;
  if (row >= N) return;
  const _Float16* xr = X + row * (long)ldx;
  float s0 = 0, s1 = 0, s2 = 0;
  for (int k = lane; k < 128; k += 32) {
    float h = (float)xr[k];
    s0 += h * w[k*3+0]; s1 += h * w[k*3+1]; s2 += h * w[k*3+2];
  }
  for (int off = 16; off; off >>= 1) {
    s0 += __shfl_down(s0, off, 32);
    s1 += __shfl_down(s1, off, 32);
    s2 += __shfl_down(s2, off, 32);
  }
  if (lane == 0) {
    float r0 = s0 + b[0], r1 = s1 + b[1], r2 = s2 + b[2];
    out[row*3+0] = 1.0f / (1.0f + __expf(-r0));
    out[row*3+1] = 1.0f / (1.0f + __expf(-r1));
    out[row*3+2] = 1.0f / (1.0f + __expf(-r2));
  }
}

// Volumetric rendering, thread per ray.
__global__ void render_kernel(const float* __restrict__ z, int S,
                              const float* __restrict__ sraw,
                              const float* __restrict__ rgb,
                              const float* __restrict__ dirs, int B,
                              float* __restrict__ o_rgb, float* __restrict__ o_disp,
                              float* __restrict__ o_acc, float* __restrict__ o_w) {
  int ray = blockIdx.x * blockDim.x + threadIdx.x;
  if (ray >= B) return;
  float dx = dirs[ray*3+0], dy = dirs[ray*3+1], dz = dirs[ray*3+2];
  float dn = sqrtf(dx*dx + dy*dy + dz*dz);
  float trans = 1.0f, acc = 0.0f, depth = 0.0f;
  float cr = 0.0f, cg = 0.0f, cb = 0.0f;
  for (int i = 0; i < S; ++i) {
    long n = (long)ray * S + i;
    float zi = z[n];
    float dist = ((i < S - 1) ? (z[n + 1] - zi) : 1e10f) * dn;
    float sg = fmaxf(0.0f, sraw[n]);
    float alpha = 1.0f - __expf(-sg * dist);
    float wt = alpha * trans;
    cr += wt * rgb[n*3+0]; cg += wt * rgb[n*3+1]; cb += wt * rgb[n*3+2];
    depth += wt * zi; acc += wt;
    if (o_w) o_w[n] = wt;
    trans *= (1.0f - alpha + 1e-10f);
  }
  float disp = acc / depth;
  bool ok = (disp > 0.0f) && (disp < 1e10f) && (acc > 0.0f);
  disp = ok ? disp : 1e10f;
  o_rgb[ray*3+0] = cr + (1.0f - acc);
  o_rgb[ray*3+1] = cg + (1.0f - acc);
  o_rgb[ray*3+2] = cb + (1.0f - acc);
  o_disp[ray] = disp;
  o_acc[ray] = acc;
}

// Inverse-CDF importance sampling + merge with coarse z (thread per ray).
__global__ void pdf_sort_kernel(const float* __restrict__ zc,
                                const float* __restrict__ wts,
                                float* __restrict__ zf, int B) {
  int ray = blockIdx.x * blockDim.x + threadIdx.x;
  if (ray >= B) return;
  const float* z = zc + (long)ray * NUM_COARSE;
  const float* W = wts + (long)ray * NUM_COARSE;
  float bins[63], w[62], cdf[63];
  for (int i = 0; i < 63; ++i) bins[i] = 0.5f * (z[i+1] + z[i]);
  float wsum = 0.0f;
  for (int i = 0; i < 62; ++i) { w[i] = W[i+1]; wsum += w[i]; }
  float pad = fmaxf(0.0f, 1e-5f - wsum);
  for (int i = 0; i < 62; ++i) w[i] += pad / 62.0f;
  wsum += pad;
  cdf[0] = 0.0f;
  float c = 0.0f;
  for (int i = 0; i < 61; ++i) { c += w[i] / wsum; cdf[i+1] = fminf(1.0f, c); }
  cdf[62] = 1.0f;
  float smp[NUM_FINE];
  const float ueps = 1.1920929e-7f;
  for (int s = 0; s < NUM_FINE; ++s) {
    float u = (float)s * (1.0f - ueps) / (float)(NUM_FINE - 1);
    int hi = 0;  for (int i = 0; i < 63; ++i)  if (u >= cdf[i]) hi = i;
    int lo = 62; for (int i = 62; i >= 0; --i) if (u <  cdf[i]) lo = i;
    float x0 = bins[hi], x1 = bins[lo];
    float d = cdf[lo] - cdf[hi];
    float t = (d > 0.0f) ? (u - cdf[hi]) / d : 0.0f;
    t = fminf(1.0f, fmaxf(0.0f, t));
    smp[s] = x0 + t * (x1 - x0);
  }
  // merge sorted coarse z (64) with samples (128, non-decreasing)
  float* o = zf + (long)ray * NTOT;
  int i = 0, j = 0, k = 0;
  while (k < NTOT) {
    bool takeZ = (j >= NUM_FINE) || (i < NUM_COARSE && z[i] <= smp[j]);
    o[k++] = takeZ ? z[i++] : smp[j++];
  }
}

// ---------------------------------------------------------------------------
// Host orchestration
// ---------------------------------------------------------------------------
struct NetWt {
  _Float16* wt[8];   // w0..w7 transposed
  _Float16* wtb;     // w_bneck transposed
  _Float16* wtc;     // w_cond transposed
};

static void prep_weights(const float* const* P, NetWt& nw, hipStream_t s) {
  // P index map (sorted pytree keys): b0..b7=0..7, b_bneck=8, b_cond=9,
  // b_rgb=10, b_sigma=11, w0..w7=12..19, w_bneck=20, w_cond=21, w_rgb=22, w_sigma=23
  const int Ks[8]    = {63, 256, 256, 256, 256, 319, 256, 256};
  const int Kpads[8] = {64, 256, 256, 256, 256, 320, 256, 256};
  for (int i = 0; i < 8; ++i) {
    long total = 256L * Kpads[i];
    wt_prep_kernel<<<(total + 255) / 256, 256, 0, s>>>(P[12 + i], nw.wt[i], Ks[i], 256, Kpads[i]);
  }
  wt_prep_kernel<<<(256L*256 + 255)/256, 256, 0, s>>>(P[20], nw.wtb, 256, 256, 256);
  wt_prep_kernel<<<(128L*288 + 255)/256, 256, 0, s>>>(P[21], nw.wtc, 283, 128, 288);
}

static void run_mlp(const float* const* P, const NetWt& nw,
                    const _Float16* enc, long N, int S,
                    _Float16* A, _Float16* Bb, const float* vdenc,
                    float* sigma_raw, float* rgbout, hipStream_t s) {
  dim3 blk(256);
  dim3 ggrid((unsigned)((N + 255) / 256));    // 8 waves * 32 rows per block
  // L0: enc(64) -> A
  gemm_wmma_kernel<<<ggrid, blk, 0, s>>>(enc, 64, nw.wt[0], 64, P[0], A, 320, N, 64, 256, 1);
  gemm_wmma_kernel<<<ggrid, blk, 0, s>>>(A, 320, nw.wt[1], 256, P[1], Bb, 320, N, 256, 256, 1);
  gemm_wmma_kernel<<<ggrid, blk, 0, s>>>(Bb, 320, nw.wt[2], 256, P[2], A, 320, N, 256, 256, 1);
  gemm_wmma_kernel<<<ggrid, blk, 0, s>>>(A, 320, nw.wt[3], 256, P[3], Bb, 320, N, 256, 256, 1);
  gemm_wmma_kernel<<<ggrid, blk, 0, s>>>(Bb, 320, nw.wt[4], 256, P[4], A, 320, N, 256, 256, 1);
  skip_fill_kernel<<<(unsigned)((N * 64 + 255) / 256), blk, 0, s>>>(A, enc, N);
  gemm_wmma_kernel<<<ggrid, blk, 0, s>>>(A, 320, nw.wt[5], 320, P[5], Bb, 320, N, 320, 256, 1);
  gemm_wmma_kernel<<<ggrid, blk, 0, s>>>(Bb, 320, nw.wt[6], 256, P[6], A, 320, N, 256, 256, 1);
  gemm_wmma_kernel<<<ggrid, blk, 0, s>>>(A, 320, nw.wt[7], 256, P[7], Bb, 320, N, 256, 256, 1);
  // heads: x = Bb
  sigma_dot_kernel<<<(unsigned)((N + 7) / 8), blk, 0, s>>>(Bb, 320, P[23], P[11], sigma_raw, N);
  gemm_wmma_kernel<<<ggrid, blk, 0, s>>>(Bb, 320, nw.wtb, 256, P[8], A, 320, N, 256, 256, 0);
  cond_fill_kernel<<<(unsigned)((N * 32 + 255) / 256), blk, 0, s>>>(A, vdenc, N, S);
  gemm_wmma_kernel<<<ggrid, blk, 0, s>>>(A, 320, nw.wtc, 288, P[9], Bb, 320, N, 288, 128, 1);
  rgb_dot_kernel<<<(unsigned)((N + 7) / 8), blk, 0, s>>>(Bb, 320, P[22], P[10], rgbout, N);
}

extern "C" void kernel_launch(void* const* d_in, const int* in_sizes, int n_in,
                              void* d_out, int out_size, void* d_ws, size_t ws_size,
                              hipStream_t stream) {
  const float* origins = (const float*)d_in[0];
  const float* dirs    = (const float*)d_in[1];
  const float* vdirs   = (const float*)d_in[2];
  int B = in_sizes[0] / 3;
  long Nc = (long)B * NUM_COARSE;
  long Nf = (long)B * NTOT;

  const float* P0[24]; const float* P1[24];
  for (int i = 0; i < 24; ++i) { P0[i] = (const float*)d_in[3 + i]; P1[i] = (const float*)d_in[27 + i]; }

  // ---- carve workspace ----
  char* p = (char*)d_ws;
  auto carve = [&](size_t bytes) -> void* {
    void* r = (void*)p; p += (bytes + 255) & ~(size_t)255; return r;
  };
  NetWt nw0, nw1;
  const int Kpads[8] = {64, 256, 256, 256, 256, 320, 256, 256};
  for (int i = 0; i < 8; ++i) nw0.wt[i] = (_Float16*)carve(256L * Kpads[i] * 2);
  nw0.wtb = (_Float16*)carve(256L * 256 * 2);
  nw0.wtc = (_Float16*)carve(128L * 288 * 2);
  for (int i = 0; i < 8; ++i) nw1.wt[i] = (_Float16*)carve(256L * Kpads[i] * 2);
  nw1.wtb = (_Float16*)carve(256L * 256 * 2);
  nw1.wtc = (_Float16*)carve(128L * 288 * 2);

  _Float16* enc_c = (_Float16*)carve(Nc * 64 * 2);
  _Float16* enc_f = (_Float16*)carve(Nf * 64 * 2);
  _Float16* bufA  = (_Float16*)carve(Nf * 320 * 2);
  _Float16* bufB  = (_Float16*)carve(Nf * 320 * 2);
  float* vdenc    = (float*)carve((long)B * 32 * 4);
  float* z_c      = (float*)carve(Nc * 4);
  float* z_f      = (float*)carve(Nf * 4);
  float* wts_c    = (float*)carve(Nc * 4);
  float* sig_raw  = (float*)carve(Nf * 4);
  float* rgb_buf  = (float*)carve(Nf * 3 * 4);

  float* out = (float*)d_out;
  float* o_rgb_c  = out;
  float* o_disp_c = out + 3L * B;
  float* o_acc_c  = out + 4L * B;
  float* o_rgb_f  = out + 5L * B;
  float* o_disp_f = out + 8L * B;
  float* o_acc_f  = out + 9L * B;

  dim3 blk(256);

  // ---- weight prep (both networks) ----
  prep_weights(P0, nw0, stream);
  prep_weights(P1, nw1, stream);

  // ---- coarse pass ----
  gen_coarse_kernel<<<(unsigned)((Nc + 255) / 256), blk, 0, stream>>>(origins, dirs, enc_c, z_c, B);
  gen_vdenc_kernel<<<(B + 255) / 256, blk, 0, stream>>>(vdirs, vdenc, B);
  run_mlp(P0, nw0, enc_c, Nc, NUM_COARSE, bufA, bufB, vdenc, sig_raw, rgb_buf, stream);
  render_kernel<<<(B + 255) / 256, blk, 0, stream>>>(z_c, NUM_COARSE, sig_raw, rgb_buf,
                                                     dirs, B, o_rgb_c, o_disp_c, o_acc_c, wts_c);

  // ---- fine sampling ----
  pdf_sort_kernel<<<(B + 63) / 64, 64, 0, stream>>>(z_c, wts_c, z_f, B);
  gen_fine_kernel<<<(unsigned)((Nf + 255) / 256), blk, 0, stream>>>(origins, dirs, z_f, enc_f, B);

  // ---- fine pass ----
  run_mlp(P1, nw1, enc_f, Nf, NTOT, bufA, bufB, vdenc, sig_raw, rgb_buf, stream);
  render_kernel<<<(B + 255) / 256, blk, 0, stream>>>(z_f, NTOT, sig_raw, rgb_buf,
                                                     dirs, B, o_rgb_f, o_disp_f, o_acc_f, nullptr);
}
